// Recursive_44040594653244
// MI455X (gfx1250) — matrix operand order repaired and checked by reading, not compiled, not run on hardware
//
#include <hip/hip_runtime.h>
#include <hip/hip_fp16.h>
#include <math.h>

typedef __attribute__((ext_vector_type(16))) _Float16 v16h;
typedef __attribute__((ext_vector_type(8)))  _Float16 v8h;
typedef __attribute__((ext_vector_type(8)))  float    v8f;

#define HIDDEN 512
#define BATCH  256
#define NSTEP  255    // reduce steps i = 1..255
#define NTILES 512    // 16 m-tiles * 32 n-tiles per step
#define NBLK   64     // persistent chain grid: 64 blocks * 8 waves = 512 waves

// ---------------------------------------------------------------------------
// Kernel 1: transpose + f32->f16 convert Wl and Wr.  WxT[n][k] = Wx[k][n].
// ---------------------------------------------------------------------------
__global__ void transpose_cvt_kernel(const float* __restrict__ Wl,
                                     const float* __restrict__ Wr,
                                     _Float16* __restrict__ WlT,
                                     _Float16* __restrict__ WrT) {
  int idx   = blockIdx.x * blockDim.x + threadIdx.x;   // 0 .. 2*512*512-1
  int which = idx >> 18;
  int r     = idx & ((1 << 18) - 1);
  int n     = r >> 9;
  int k     = r & 511;
  const float* src = which ? Wr : Wl;
  _Float16*    dst = which ? WrT : WlT;
  dst[(size_t)n * HIDDEN + k] = (_Float16)src[(size_t)k * HIDDEN + n];
}

// ---------------------------------------------------------------------------
// Kernel 2: h0 = emb[tokens[0, b]]  (f16 A operand for step 1)
// ---------------------------------------------------------------------------
__global__ void init_h0_kernel(const int* __restrict__ tokens,
                               const float* __restrict__ emb,
                               _Float16* __restrict__ H0) {
  int idx = blockIdx.x * blockDim.x + threadIdx.x;     // 0 .. 256*512-1
  int b   = idx >> 9;
  int k   = idx & 511;
  int tok = tokens[b];                                 // row 0 of (511,256)
  H0[idx] = (_Float16)emb[(size_t)tok * HIDDEN + k];
}

// ---------------------------------------------------------------------------
// Kernel 3: zero the device-barrier counters (arrival, generation).
// ---------------------------------------------------------------------------
__global__ void sync_init_kernel(unsigned* __restrict__ s) {
  if (threadIdx.x < 2) s[threadIdx.x] = 0u;
}

// ---------------------------------------------------------------------------
// Kernel 4 (parallel bulk): R = emb[tok]@Wr + bias, stored in C-fragment
// layout: R[(step-1)][tile][lane][0..7] so the chain kernel reads one b128.
// ---------------------------------------------------------------------------
__global__ void __launch_bounds__(256)
r_gemm_kernel(const int* __restrict__ tokens,
              const float* __restrict__ emb,
              const _Float16* __restrict__ WrT,
              const float* __restrict__ bias,
              _Float16* __restrict__ R) {
  const int lane  = threadIdx.x & 31;
  const int wave  = threadIdx.x >> 5;
  const int tile  = blockIdx.x * 8 + wave;             // 0 .. 255*512-1
  const int i     = (tile >> 9) + 1;                   // step 1..255
  const int rem   = tile & (NTILES - 1);               // tile within step
  const int mtile = rem >> 5;
  const int ntile = rem & 31;
  const int lhalf = lane >> 4;
  const int lmod  = lane & 15;

  const int m   = mtile * 16 + lmod;
  const int tok = tokens[(2 * i - 1) * BATCH + m];
  const float* arow = emb + (size_t)tok * HIDDEN;
  const _Float16* brow = WrT + (size_t)(ntile * 16 + lmod) * HIDDEN;

  v8f acc = {};
#pragma unroll 4
  for (int k0 = 0; k0 < HIDDEN; k0 += 32) {
    const int kA0 = k0 + lhalf * 8;
    const int kA1 = kA0 + 16;
    float4 f0 = *(const float4*)(arow + kA0);
    float4 f1 = *(const float4*)(arow + kA0 + 4);
    float4 f2 = *(const float4*)(arow + kA1);
    float4 f3 = *(const float4*)(arow + kA1 + 4);
    v16h a;
    a[0]=(_Float16)f0.x;  a[1]=(_Float16)f0.y;  a[2]=(_Float16)f0.z;  a[3]=(_Float16)f0.w;
    a[4]=(_Float16)f1.x;  a[5]=(_Float16)f1.y;  a[6]=(_Float16)f1.z;  a[7]=(_Float16)f1.w;
    a[8]=(_Float16)f2.x;  a[9]=(_Float16)f2.y;  a[10]=(_Float16)f2.z; a[11]=(_Float16)f2.w;
    a[12]=(_Float16)f3.x; a[13]=(_Float16)f3.y; a[14]=(_Float16)f3.z; a[15]=(_Float16)f3.w;

    const int kB = k0 + lhalf * 8;
    v8h b0 = *(const v8h*)(brow + kB);
    v8h b1 = *(const v8h*)(brow + kB + 16);
    v16h b = __builtin_shufflevector(b0, b1, 0,1,2,3,4,5,6,7,8,9,10,11,12,13,14,15);

    acc = __builtin_amdgcn_wmma_f32_16x16x32_f16(false, a, false, b,
                                                 (short)0, acc, false, false);
  }

  const float bn = bias[ntile * 16 + lmod];
  v8h rv;
#pragma unroll
  for (int j = 0; j < 8; ++j) rv[j] = (_Float16)(acc[j] + bn);
  // fragment-layout store: one contiguous b128 per lane
  *(v8h*)(R + ((((size_t)(i - 1) * NTILES + rem) * 32) + lane) * 8) = rv;
}

// ---------------------------------------------------------------------------
// Kernel 5 (persistent sequential chain): runs all 255 dependent steps in one
// dispatch.  Each wave owns one 16x16 tile; Wl B-fragments live in VGPRs for
// the whole kernel.  Device-scope generation barrier between steps.
// ---------------------------------------------------------------------------
__global__ void __launch_bounds__(256)
chain_kernel(const _Float16* __restrict__ R,
             const _Float16* __restrict__ WlT,
             _Float16* __restrict__ H0,
             _Float16* __restrict__ H1,
             float* __restrict__ outf,
             unsigned* __restrict__ sync) {   // sync[0]=arrivals, sync[1]=generation
  const int lane  = threadIdx.x & 31;
  const int wave  = threadIdx.x >> 5;
  const int tile  = blockIdx.x * 8 + wave;             // 0..511
  const int mtile = tile >> 5;
  const int ntile = tile & 31;
  const int lhalf = lane >> 4;
  const int lmod  = lane & 15;

  // Preload this tile's Wl column fragments: invariant across all 255 steps.
  const _Float16* brow = WlT + (size_t)(ntile * 16 + lmod) * HIDDEN;
  v16h bfrag[16];
#pragma unroll
  for (int kk = 0; kk < 16; ++kk) {
    const int kB = kk * 32 + lhalf * 8;
    v8h b0 = *(const v8h*)(brow + kB);
    v8h b1 = *(const v8h*)(brow + kB + 16);
    bfrag[kk] = __builtin_shufflevector(b0, b1, 0,1,2,3,4,5,6,7,8,9,10,11,12,13,14,15);
  }

  const size_t arow_off = (size_t)(mtile * 16 + lmod) * HIDDEN;
  _Float16* bufs[2] = {H0, H1};

  for (int i = 1; i <= NSTEP; ++i) {
    const _Float16* arow = bufs[(i - 1) & 1] + arow_off;
    _Float16*       Hout = bufs[i & 1];

    const _Float16* rptr = R + ((((size_t)(i - 1) * NTILES + tile) * 32) + lane) * 8;
    if (i < NSTEP)  // prefetch next step's R fragment while we compute
      __builtin_prefetch(rptr + (size_t)NTILES * 32 * 8, 0, 1);

    v8f acc = {};
#pragma unroll
    for (int kk = 0; kk < 16; ++kk) {
      const int kA0 = kk * 32 + lhalf * 8;
      v8h a0 = *(const v8h*)(arow + kA0);
      v8h a1 = *(const v8h*)(arow + kA0 + 16);
      v16h a = __builtin_shufflevector(a0, a1, 0,1,2,3,4,5,6,7,8,9,10,11,12,13,14,15);
      acc = __builtin_amdgcn_wmma_f32_16x16x32_f16(false, a, false, bfrag[kk],
                                                   (short)0, acc, false, false);
    }

    const v8h rv = *(const v8h*)rptr;       // bias already folded into R
    const int n = ntile * 16 + lmod;
#pragma unroll
    for (int j = 0; j < 8; ++j) {
      const int m = mtile * 16 + lhalf * 8 + j;
      float h = tanhf(acc[j] + (float)rv[j]);
      Hout[(size_t)m * HIDDEN + n] = (_Float16)h;
      if (i == NSTEP) outf[(size_t)m * HIDDEN + n] = h;
    }

    if (i < NSTEP) {
      // ---- device-wide generation barrier ----
      __threadfence();                       // release this wave's H stores
      __syncthreads();
      if (threadIdx.x == 0) {
        unsigned prev = __hip_atomic_fetch_add(&sync[0], 1u, __ATOMIC_ACQ_REL,
                                               __HIP_MEMORY_SCOPE_AGENT);
        if (prev == NBLK - 1) {
          __hip_atomic_store(&sync[0], 0u, __ATOMIC_RELAXED, __HIP_MEMORY_SCOPE_AGENT);
          __hip_atomic_fetch_add(&sync[1], 1u, __ATOMIC_RELEASE, __HIP_MEMORY_SCOPE_AGENT);
        } else {
          while (__hip_atomic_load(&sync[1], __ATOMIC_ACQUIRE,
                                   __HIP_MEMORY_SCOPE_AGENT) < (unsigned)i)
            __builtin_amdgcn_s_sleep(1);
        }
      }
      __syncthreads();
      __threadfence();                       // acquire: drop stale lines before reload
    }
  }
}

// ---------------------------------------------------------------------------
extern "C" void kernel_launch(void* const* d_in, const int* in_sizes, int n_in,
                              void* d_out, int out_size, void* d_ws, size_t ws_size,
                              hipStream_t stream) {
  const int*   tokens = (const int*)d_in[0];   // (511, 256) int32
  const float* emb    = (const float*)d_in[1]; // (32000, 512)
  const float* Wl     = (const float*)d_in[2]; // (512, 512)
  const float* Wr     = (const float*)d_in[3]; // (512, 512)
  const float* bias   = (const float*)d_in[4]; // (512,)
  float*       out    = (float*)d_out;         // (256, 512)

  char* ws = (char*)d_ws;
  const size_t W_BYTES = (size_t)HIDDEN * HIDDEN * 2;            // 512 KB
  const size_t H_BYTES = (size_t)BATCH * HIDDEN * 2;             // 256 KB
  const size_t R_BYTES = (size_t)NSTEP * NTILES * 32 * 8 * 2;    // ~66.8 MB
  _Float16* WlT  = (_Float16*)(ws);
  _Float16* WrT  = (_Float16*)(ws + W_BYTES);
  _Float16* H0   = (_Float16*)(ws + 2 * W_BYTES);
  _Float16* H1   = (_Float16*)(ws + 2 * W_BYTES + H_BYTES);
  _Float16* R    = (_Float16*)(ws + 2 * W_BYTES + 2 * H_BYTES);
  unsigned* sync = (unsigned*)(ws + 2 * W_BYTES + 2 * H_BYTES + R_BYTES);

  transpose_cvt_kernel<<<(2 * 512 * 512) / 256, 256, 0, stream>>>(Wl, Wr, WlT, WrT);
  init_h0_kernel<<<(BATCH * HIDDEN) / 256, 256, 0, stream>>>(tokens, emb, H0);
  sync_init_kernel<<<1, 32, 0, stream>>>(sync);
  r_gemm_kernel<<<(NSTEP * NTILES) / 8, 256, 0, stream>>>(tokens, emb, WrT, bias, R);
  chain_kernel<<<NBLK, 256, 0, stream>>>(R, WlT, H0, H1, out, sync);
}
